// WordSAGE_4518305595690
// MI455X (gfx1250) — matrix operand order repaired and checked by reading, not compiled, not run on hardware
//
#include <hip/hip_runtime.h>
#include <hip/hip_bf16.h>

#define GN 10000
#define GE 160000
#define GD 2500
#define GC 16

typedef __attribute__((ext_vector_type(16))) __bf16 v16bf;
typedef __attribute__((ext_vector_type(8)))  float  v8f;

__device__ __forceinline__ unsigned short f2bf(float x) {
    unsigned u = __float_as_uint(x);
    unsigned r = (u + 0x7FFFu + ((u >> 16) & 1u)) >> 16;
    return (unsigned short)r;
}

// ---------------- small utility kernels ----------------

__global__ void k_zero_i32(int* p, int n) {
    int i = blockIdx.x * blockDim.x + threadIdx.x;
    if (i < n) p[i] = 0;
}

__global__ void k_count(const long long* __restrict__ dst, int* __restrict__ deg, int e) {
    int i = blockIdx.x * blockDim.x + threadIdx.x;
    if (i < e) atomicAdd(&deg[(int)dst[i]], 1);
}

// single-block exclusive scan of deg[0..n) -> offs[0..n], also copies to cursor
__global__ __launch_bounds__(1024) void k_scan(const int* __restrict__ deg,
                                               int* __restrict__ offs,
                                               int* __restrict__ cursor, int n) {
    __shared__ int sums[1024];
    int t = threadIdx.x;
    int per = (n + 1023) / 1024;
    int start = t * per;
    int s = 0;
    for (int i = 0; i < per; ++i) { int idx = start + i; if (idx < n) s += deg[idx]; }
    sums[t] = s;
    __syncthreads();
    for (int off = 1; off < 1024; off <<= 1) {
        int v = (t >= off) ? sums[t - off] : 0;
        __syncthreads();
        sums[t] += v;
        __syncthreads();
    }
    int run = (t == 0) ? 0 : sums[t - 1];
    for (int i = 0; i < per; ++i) {
        int idx = start + i;
        if (idx < n) { offs[idx] = run; cursor[idx] = run; run += deg[idx]; }
    }
    if (t == 1023) offs[n] = sums[1023];
}

__global__ void k_fill(const long long* __restrict__ ei, int* __restrict__ cursor,
                       int* __restrict__ esrc, int e) {
    int i = blockIdx.x * blockDim.x + threadIdx.x;
    if (i < e) {
        int d = (int)ei[(size_t)e + i];
        int pos = atomicAdd(&cursor[d], 1);
        esrc[pos] = (int)ei[i];
    }
}

__global__ void k_f2bf(const float* __restrict__ in, unsigned short* __restrict__ out, long long n) {
    long long i = (long long)blockIdx.x * blockDim.x + threadIdx.x;
    long long stride = (long long)gridDim.x * blockDim.x;
    for (; i < n; i += stride) out[i] = f2bf(in[i]);
}

// CSR gather-mean over in-neighbors, fused fp32 -> bf16
__global__ void k_gather_mean_bf16(const float* __restrict__ h, const int* __restrict__ offs,
                                   const int* __restrict__ esrc, unsigned short* __restrict__ outb,
                                   int d) {
    int n = blockIdx.x;
    int f = blockIdx.y * blockDim.x + threadIdx.x;
    if (f >= d) return;
    int beg = offs[n], end = offs[n + 1];
    float s = 0.0f;
    for (int i = beg; i < end; ++i)
        s += h[(size_t)esrc[i] * d + f];
    int cnt = end - beg;
    float m = s / (float)(cnt > 0 ? cnt : 1);
    outb[(size_t)n * d + f] = f2bf(m);
}

// ---------------- WMMA bf16 GEMM ----------------
// C[M,N] = act( A1*B1 (+ A2*B2) + bias ),  A row-major [M,K] bf16, B row-major [K,N] bf16
// Block: 128 threads = 4 waves (2x2), block tile 128x128, wave tile 64x64, K-step 32.
// LDS double-buffered; B staged transposed via 4x4 register transpose.
#define TILE_M 128
#define TILE_N 128
#define TILE_K 32
#define LDA 40
#define LDB 40

__device__ __forceinline__ void stage_tiles(unsigned short* __restrict__ sAb,
                                            unsigned short* __restrict__ sBb,
                                            const unsigned short* __restrict__ A,
                                            const unsigned short* __restrict__ B,
                                            int m0, int n0, int kk,
                                            int M, int N, int K, int tid) {
    // ---- A tile: 128 rows x 32 cols, one row per thread ----
    {
        int r = tid;
        int grow = m0 + r;
        unsigned short* dstp = &sAb[r * LDA];
        if (grow < M && kk + TILE_K <= K) {
            const unsigned short* srcp = &A[(size_t)grow * K + kk];
#pragma unroll
            for (int j = 0; j < 8; ++j)
                *(uint2*)(dstp + j * 4) = *(const uint2*)(srcp + j * 4);
        } else {
            for (int j = 0; j < TILE_K; ++j) {
                int gc = kk + j;
                dstp[j] = (grow < M && gc < K) ? A[(size_t)grow * K + gc] : (unsigned short)0;
            }
        }
    }
    // ---- B tile transposed: sB[n][k] = B[kk+k][n0+n]; 4x4 register transpose ----
    union U2 { uint2 v; unsigned short s[4]; };
#pragma unroll
    for (int bi = 0; bi < 2; ++bi) {
        int b  = tid + bi * 128;       // 256 4x4 blocks: 8 k-blocks x 32 n-blocks
        int kb = b >> 5;               // 0..7
        int nb = b & 31;               // 0..31
        int gk0 = kk + kb * 4;
        int gn0 = n0 + nb * 4;
        if (gk0 + 3 < K && gn0 + 3 < N) {
            U2 row[4];
#pragma unroll
            for (int r = 0; r < 4; ++r)
                row[r].v = *(const uint2*)&B[(size_t)(gk0 + r) * N + gn0];
#pragma unroll
            for (int c = 0; c < 4; ++c) {
                uint2 o;
                o.x = (unsigned)row[0].s[c] | ((unsigned)row[1].s[c] << 16);
                o.y = (unsigned)row[2].s[c] | ((unsigned)row[3].s[c] << 16);
                *(uint2*)&sBb[(nb * 4 + c) * LDB + kb * 4] = o;
            }
        } else {
#pragma unroll
            for (int c = 0; c < 4; ++c)
                for (int r = 0; r < 4; ++r) {
                    int gk = gk0 + r, gn = gn0 + c;
                    sBb[(nb * 4 + c) * LDB + kb * 4 + r] =
                        (gk < K && gn < N) ? B[(size_t)gk * N + gn] : (unsigned short)0;
                }
        }
    }
}

__global__ __launch_bounds__(128, 1)
void k_gemm(const unsigned short* __restrict__ A1, const unsigned short* __restrict__ B1,
            const unsigned short* __restrict__ A2, const unsigned short* __restrict__ B2,
            const float* __restrict__ bias, float* __restrict__ C,
            int M, int N, int K, int relu, int npairs) {
    __shared__ unsigned short sA[2][TILE_M * LDA];   // 2 x 10240 B
    __shared__ unsigned short sB[2][TILE_N * LDB];   // 2 x 10240 B

    const int tid  = threadIdx.x;
    const int lane = tid & 31;
    const int wv   = tid >> 5;        // 0..3
    const int wm   = (wv >> 1) * 64;  // wave row offset in block tile
    const int wn   = (wv & 1) * 64;   // wave col offset in block tile
    const int m0   = blockIdx.y * TILE_M;
    const int n0   = blockIdx.x * TILE_N;
    const int laneM = lane & 15;
    const int laneH = lane >> 4;      // 0/1

    v8f acc[4][4];
#pragma unroll
    for (int mi = 0; mi < 4; ++mi)
#pragma unroll
        for (int ni = 0; ni < 4; ++ni)
            acc[mi][ni] = (v8f){0.f,0.f,0.f,0.f,0.f,0.f,0.f,0.f};

    const int ktiles = (K + TILE_K - 1) / TILE_K;
    const int steps  = ktiles * npairs;

    // prologue: stage step 0 into buffer 0
    {
        int kt = 0, p = 0;
        const unsigned short* A = p ? A2 : A1;
        const unsigned short* B = p ? B2 : B1;
        stage_tiles(sA[0], sB[0], A, B, m0, n0, kt * TILE_K, M, N, K, tid);
    }
    __syncthreads();

    for (int s = 0; s < steps; ++s) {
        const int cur = s & 1;
        // stage next step into the other buffer
        if (s + 1 < steps) {
            int s1 = s + 1;
            int kt = (npairs == 2) ? (s1 >> 1) : s1;
            int p  = (npairs == 2) ? (s1 & 1) : 0;
            const unsigned short* A = p ? A2 : A1;
            const unsigned short* B = p ? B2 : B1;
            stage_tiles(sA[cur ^ 1], sB[cur ^ 1], A, B, m0, n0, kt * TILE_K, M, N, K, tid);
        }
        // compute from current buffer
        union Frag { v16bf v; uint4 q[2]; };
        Frag a[4], b[4];
#pragma unroll
        for (int ni = 0; ni < 4; ++ni) {
            int n = wn + ni * 16 + laneM;
            const unsigned short* sp = &sB[cur][n * LDB];
            b[ni].q[0] = *(const uint4*)(sp + laneH * 8);
            b[ni].q[1] = *(const uint4*)(sp + 16 + laneH * 8);
        }
#pragma unroll
        for (int mi = 0; mi < 4; ++mi) {
            int m = wm + mi * 16 + laneM;
            const unsigned short* sp = &sA[cur][m * LDA];
            a[mi].q[0] = *(const uint4*)(sp + laneH * 8);
            a[mi].q[1] = *(const uint4*)(sp + 16 + laneH * 8);
        }
#pragma unroll
        for (int mi = 0; mi < 4; ++mi)
#pragma unroll
            for (int ni = 0; ni < 4; ++ni)
                acc[mi][ni] = __builtin_amdgcn_wmma_f32_16x16x32_bf16(
                    false, a[mi].v, false, b[ni].v, (short)0, acc[mi][ni], false, false);
        __syncthreads();
    }

    // ---- epilogue: bias + relu + bounds-checked store ----
#pragma unroll
    for (int mi = 0; mi < 4; ++mi) {
#pragma unroll
        for (int ni = 0; ni < 4; ++ni) {
            int col = n0 + wn + ni * 16 + laneM;
            if (col >= N) continue;
            float bv = bias ? bias[col] : 0.0f;
#pragma unroll
            for (int i = 0; i < 8; ++i) {
                int row = m0 + wm + mi * 16 + laneH * 8 + i;
                if (row < M) {
                    float v = acc[mi][ni][i] + bv;
                    if (relu) v = v > 0.0f ? v : 0.0f;
                    C[(size_t)row * N + col] = v;
                }
            }
        }
    }
}

// final small layer: out[N,C] = h @ Wc2 + bc2 (fp32, memory-bound)
__global__ void k_out(const float* __restrict__ h, const float* __restrict__ W,
                      const float* __restrict__ b, float* __restrict__ out) {
    int idx = blockIdx.x * blockDim.x + threadIdx.x;
    if (idx >= GN * GC) return;
    int n = idx / GC, c = idx % GC;
    const float* hr = h + (size_t)n * GD;
    float s = b[c];
    for (int k = 0; k < GD; ++k) s += hr[k] * W[(size_t)k * GC + c];
    out[idx] = s;
}

// ---------------- host ----------------

extern "C" void kernel_launch(void* const* d_in, const int* in_sizes, int n_in,
                              void* d_out, int out_size, void* d_ws, size_t ws_size,
                              hipStream_t stream) {
    (void)in_sizes; (void)n_in; (void)out_size; (void)ws_size;
    const float*     x   = (const float*)d_in[0];
    const long long* ei  = (const long long*)d_in[1];
    const float* W1s = (const float*)d_in[2];
    const float* W1n = (const float*)d_in[3];
    const float* b1  = (const float*)d_in[4];
    const float* W2s = (const float*)d_in[5];
    const float* W2n = (const float*)d_in[6];
    const float* b2  = (const float*)d_in[7];
    const float* Wc1 = (const float*)d_in[8];
    const float* bc1 = (const float*)d_in[9];
    const float* Wc2 = (const float*)d_in[10];
    const float* bc2 = (const float*)d_in[11];
    float* out = (float*)d_out;

    // workspace layout
    char* ws = (char*)d_ws;
    size_t off = 0;
    auto take = [&](size_t bytes) { char* p = ws + off; off = (off + bytes + 255) & ~(size_t)255; return p; };
    int* deg    = (int*)take((size_t)GN * 4);
    int* offs   = (int*)take((size_t)(GN + 1) * 4);
    int* cursor = (int*)take((size_t)GN * 4);
    int* esrc   = (int*)take((size_t)GE * 4);
    unsigned short* Xb = (unsigned short*)take((size_t)GN * GD * 2);
    unsigned short* Mb = (unsigned short*)take((size_t)GN * GD * 2);
    unsigned short* Wa = (unsigned short*)take((size_t)GD * GD * 2);
    unsigned short* Wb = (unsigned short*)take((size_t)GD * GD * 2);
    float* H1 = (float*)take((size_t)GN * GD * 4);
    float* H2 = (float*)take((size_t)GN * GD * 4);

    const long long ND = (long long)GN * GD;
    const long long DD = (long long)GD * GD;
    dim3 gemmGrid((GD + TILE_N - 1) / TILE_N, (GN + TILE_M - 1) / TILE_M);
    dim3 meanGrid(GN, (GD + 255) / 256);

    // build CSR (by dst)
    k_zero_i32<<<(GN + 255) / 256, 256, 0, stream>>>(deg, GN);
    k_count<<<(GE + 255) / 256, 256, 0, stream>>>(ei + GE, deg, GE);
    k_scan<<<1, 1024, 0, stream>>>(deg, offs, cursor, GN);
    k_fill<<<(GE + 255) / 256, 256, 0, stream>>>(ei, cursor, esrc, GE);

    // ---- layer 1 ----
    k_f2bf<<<2048, 256, 0, stream>>>(x, Xb, ND);
    k_gather_mean_bf16<<<meanGrid, 256, 0, stream>>>(x, offs, esrc, Mb, GD);
    k_f2bf<<<2048, 256, 0, stream>>>(W1s, Wa, DD);
    k_f2bf<<<2048, 256, 0, stream>>>(W1n, Wb, DD);
    k_gemm<<<gemmGrid, 128, 0, stream>>>(Xb, Wa, Mb, Wb, b1, H1, GN, GD, GD, 1, 2);

    // ---- layer 2 ----
    k_gather_mean_bf16<<<meanGrid, 256, 0, stream>>>(H1, offs, esrc, Mb, GD);
    k_f2bf<<<2048, 256, 0, stream>>>(H1, Xb, ND);
    k_f2bf<<<2048, 256, 0, stream>>>(W2s, Wa, DD);
    k_f2bf<<<2048, 256, 0, stream>>>(W2n, Wb, DD);
    k_gemm<<<gemmGrid, 128, 0, stream>>>(Xb, Wa, Mb, Wb, b2, H2, GN, GD, GD, 1, 2);

    // ---- classifier hidden: H1 = relu(H2 @ Wc1 + bc1) ----
    k_f2bf<<<2048, 256, 0, stream>>>(H2, Xb, ND);
    k_f2bf<<<2048, 256, 0, stream>>>(Wc1, Wa, DD);
    k_gemm<<<gemmGrid, 128, 0, stream>>>(Xb, Wa, (const unsigned short*)nullptr,
                                         (const unsigned short*)nullptr, bc1, H1, GN, GD, GD, 1, 1);

    // ---- final: out = H1 @ Wc2 + bc2 ----
    k_out<<<(GN * GC + 255) / 256, 256, 0, stream>>>(H1, Wc2, bc2, out);
}
